// Classifier_Model_67654324847284
// MI455X (gfx1250) — matrix operand order repaired and checked by reading, not compiled, not run on hardware
//
#include <hip/hip_runtime.h>
#include <hip/hip_bf16.h>
#include <math.h>

typedef __attribute__((ext_vector_type(16))) _Float16 v16h;
typedef __attribute__((ext_vector_type(8)))  _Float16 v8h;
typedef __attribute__((ext_vector_type(8)))  float    v8f;

union V16H { v16h v; v8h h[2]; };

#define MTOT   40960      // B*T
#define BATCH  2048
#define TSEQ   20
#define DIN    2048
#define HID    1024
#define EDIM   256
#define NHEAD  4
#define HD     64
#define NCLS   1000
#define NCLSP  1024

// ---------------------------------------------------------------- utilities
__device__ inline float block_sum256(float v, float* red) {
    // blockDim.x == 256, 8 waves of 32
    #pragma unroll
    for (int o = 16; o > 0; o >>= 1) v += __shfl_xor(v, o, 32);
    const int wave = threadIdx.x >> 5;
    const int lane = threadIdx.x & 31;
    if (lane == 0) red[wave] = v;
    __syncthreads();
    float t = (threadIdx.x < 8) ? red[threadIdx.x] : 0.0f;
    if (wave == 0) {
        #pragma unroll
        for (int o = 16; o > 0; o >>= 1) t += __shfl_xor(t, o, 32);
        if (lane == 0) red[0] = t;
    }
    __syncthreads();
    float r = red[0];
    __syncthreads();
    return r;
}

__device__ inline float gelu_exact(float x) {
    return 0.5f * x * (1.0f + erff(x * 0.70710678118654752f));
}

// ---------------------------------------------------------------- f32 -> f16
__global__ __launch_bounds__(256)
void cvt_f32_f16(const float* __restrict__ src, _Float16* __restrict__ dst, size_t n) {
    size_t i = (size_t)blockIdx.x * 256 + threadIdx.x;
    if (i < n) dst[i] = (_Float16)src[i];
}

// ---------------------------------------------------------------- WMMA GEMM
// C[M,Npad] (f32, ldc, store only n<Nstore) = alpha * A[M,K](f16,row-major) x W[N,K]^T + bias
// block = 128 threads (4 waves); wave w -> m-tile (4*by+w); block x -> 4 n-tiles
//
// Operand layouts used here (A/B contiguous k order is this code's choice, matched on both sides):
//  A 16x32 f16 : lane l holds row M=l&15; reg elems e=0..15 -> K=kh+e,
//                with kh = (l>=16)?16:0  (contiguous run of 16).
//  B 32x16 f16 : lane l holds col N=l&15; reg elems e=0..15 -> K=kb+e,
//                with kb = (l>=16)?16:0  (contiguous run of 16).
//  C/D 16x16 f32: lane l col N=l&15; elem r -> row M=r + ((l>=16)?8:0).
__global__ __launch_bounds__(128)
void gemm_f16_wmma(const _Float16* __restrict__ A, int lda,
                   const _Float16* __restrict__ W, int ldw,
                   const float* __restrict__ bias,
                   float* __restrict__ C, int ldc,
                   int M, int K, int Nstore, float alpha)
{
    const int lane  = threadIdx.x & 31;
    const int wave  = threadIdx.x >> 5;
    const int mTile = (blockIdx.y * 4 + wave) * 16;
    const int nTile0 = blockIdx.x * 64;
    if (mTile >= M) return;

    const int r  = lane & 15;
    const int hi = lane >> 4;          // 0 for lanes 0-15, 1 for lanes 16-31
    const int kHalfA = hi ? 16 : 0;    // A-operand k offset (run of 16)
    const int kHalfB = hi ? 16 : 0;    // B-operand k offset (run of 16)

    v8f acc0 = {}, acc1 = {}, acc2 = {}, acc3 = {};
    const _Float16* aRow = A + (size_t)(mTile + r) * lda + kHalfA;
    const _Float16* wRow = W + (size_t)(nTile0 + r) * ldw + kHalfB;
    const size_t wStride = (size_t)16 * ldw;

    for (int k = 0; k < K; k += 32) {
        V16H a;
        a.h[0] = *(const v8h*)(aRow + k);
        a.h[1] = *(const v8h*)(aRow + k + 8);
        V16H b0, b1, b2, b3;
        const _Float16* w0 = wRow + k;
        b0.h[0] = *(const v8h*)(w0);
        b0.h[1] = *(const v8h*)(w0 + 8);
        b1.h[0] = *(const v8h*)(w0 + wStride);
        b1.h[1] = *(const v8h*)(w0 + wStride + 8);
        b2.h[0] = *(const v8h*)(w0 + 2 * wStride);
        b2.h[1] = *(const v8h*)(w0 + 2 * wStride + 8);
        b3.h[0] = *(const v8h*)(w0 + 3 * wStride);
        b3.h[1] = *(const v8h*)(w0 + 3 * wStride + 8);
        acc0 = __builtin_amdgcn_wmma_f32_16x16x32_f16(false, a.v, false, b0.v, (short)0, acc0, false, false);
        acc1 = __builtin_amdgcn_wmma_f32_16x16x32_f16(false, a.v, false, b1.v, (short)0, acc1, false, false);
        acc2 = __builtin_amdgcn_wmma_f32_16x16x32_f16(false, a.v, false, b2.v, (short)0, acc2, false, false);
        acc3 = __builtin_amdgcn_wmma_f32_16x16x32_f16(false, a.v, false, b3.v, (short)0, acc3, false, false);
    }

    const int colBase = nTile0 + (lane & 15);
    const int rowBase = mTile + (hi ? 8 : 0);
    v8f accs[4] = {acc0, acc1, acc2, acc3};
    #pragma unroll
    for (int j = 0; j < 4; ++j) {
        const int col = colBase + j * 16;
        if (col >= Nstore) continue;
        const float bv = bias ? bias[col] : 0.0f;
        #pragma unroll
        for (int i = 0; i < 8; ++i) {
            C[(size_t)(rowBase + i) * ldc + col] = accs[j][i] * alpha + bv;
        }
    }
}

// ---------------------------------------------------------------- LN + GELU (projector stage 1), N=1024
__global__ __launch_bounds__(256)
void ln_gelu_f16(const float* __restrict__ X, const float* __restrict__ g,
                 const float* __restrict__ bta, _Float16* __restrict__ Y, int N)
{
    __shared__ float red[8];
    const size_t row = blockIdx.x;
    const float* x = X + row * N;
    _Float16*    y = Y + row * N;
    float s = 0.0f;
    for (int i = threadIdx.x; i < N; i += 256) s += x[i];
    const float mean = block_sum256(s, red) / (float)N;
    float s2 = 0.0f;
    for (int i = threadIdx.x; i < N; i += 256) { float d = x[i] - mean; s2 += d * d; }
    const float inv = rsqrtf(block_sum256(s2, red) / (float)N + 1e-6f);
    for (int i = threadIdx.x; i < N; i += 256) {
        float v = (x[i] - mean) * inv * g[i] + bta[i];
        y[i] = (_Float16)gelu_exact(v);
    }
}

// ---------------------------------------------------------------- LN + sincos PE (projector stage 2), N=256
__global__ __launch_bounds__(256)
void ln_pe_f32f16(const float* __restrict__ X, const float* __restrict__ g,
                  const float* __restrict__ bta,
                  float* __restrict__ R, _Float16* __restrict__ Hh)
{
    __shared__ float red[8];
    const size_t row = blockIdx.x;
    const int c = threadIdx.x;
    const float v = X[row * EDIM + c];
    const float mean = block_sum256(v, red) * (1.0f / EDIM);
    const float d = v - mean;
    const float inv = rsqrtf(block_sum256(d * d, red) * (1.0f / EDIM) + 1e-6f);
    float y = d * inv * g[c] + bta[c];
    const int t = (int)(row % TSEQ);
    const int p = c >> 1;
    // div = exp((2p) * (-ln(10000)/256))
    const float ang = (float)t * expf((float)(2 * p) * (-9.210340371976184f / 256.0f));
    y += (c & 1) ? cosf(ang) : sinf(ang);
    R[row * EDIM + c]  = y;
    Hh[row * EDIM + c] = (_Float16)y;
}

// ---------------------------------------------------------------- post-norm residual LN, N=256 (in-place on R)
__global__ __launch_bounds__(256)
void residual_ln(float* __restrict__ R, const float* __restrict__ Dl,
                 const float* __restrict__ g, const float* __restrict__ bta,
                 _Float16* __restrict__ Hh)
{
    __shared__ float red[8];
    const size_t row = blockIdx.x;
    const int c = threadIdx.x;
    const float v = R[row * EDIM + c] + Dl[row * EDIM + c];
    const float mean = block_sum256(v, red) * (1.0f / EDIM);
    const float d = v - mean;
    const float inv = rsqrtf(block_sum256(d * d, red) * (1.0f / EDIM) + 1e-6f);
    const float y = d * inv * g[c] + bta[c];
    R[row * EDIM + c]  = y;
    Hh[row * EDIM + c] = (_Float16)y;
}

// ---------------------------------------------------------------- fused causal attention, one block per (batch, head)
__global__ __launch_bounds__(128)
void attention_fused(const float* __restrict__ qkv, _Float16* __restrict__ attn)
{
    const int bh = blockIdx.x;
    const int b  = bh / NHEAD;
    const int h  = bh % NHEAD;
    __shared__ float q[TSEQ * HD], k[TSEQ * HD], v[TSEQ * HD];
    __shared__ float sc[TSEQ * (TSEQ + 1)];

    const float* base = qkv + (size_t)b * TSEQ * (3 * EDIM) + h * HD;
    for (int i = threadIdx.x; i < TSEQ * HD; i += 128) {
        const int t = i / HD, d = i % HD;
        const float* rp = base + (size_t)t * (3 * EDIM) + d;
        q[i] = rp[0];
        k[i] = rp[EDIM];
        v[i] = rp[2 * EDIM];
    }
    __syncthreads();

    const float scale = 0.125f; // 1/sqrt(64)
    for (int e = threadIdx.x; e < TSEQ * TSEQ; e += 128) {
        const int t = e / TSEQ, s = e % TSEQ;
        float a;
        if (s <= t) {
            a = 0.0f;
            #pragma unroll 8
            for (int d = 0; d < HD; ++d) a += q[t * HD + d] * k[s * HD + d];
            a *= scale;
        } else {
            a = -__builtin_inff();
        }
        sc[t * (TSEQ + 1) + s] = a;
    }
    __syncthreads();

    if (threadIdx.x < TSEQ) {
        const int t = threadIdx.x;
        float mx = -__builtin_inff();
        for (int s = 0; s <= t; ++s) mx = fmaxf(mx, sc[t * (TSEQ + 1) + s]);
        float sum = 0.0f;
        for (int s = 0; s <= t; ++s) {
            const float p = expf(sc[t * (TSEQ + 1) + s] - mx);
            sc[t * (TSEQ + 1) + s] = p;
            sum += p;
        }
        const float invs = 1.0f / sum;
        for (int s = 0; s <= t; ++s) sc[t * (TSEQ + 1) + s] *= invs;
    }
    __syncthreads();

    for (int i = threadIdx.x; i < TSEQ * HD; i += 128) {
        const int t = i / HD, d = i % HD;
        float a = 0.0f;
        for (int s = 0; s <= t; ++s) a += sc[t * (TSEQ + 1) + s] * v[s * HD + d];
        attn[((size_t)b * TSEQ + t) * EDIM + h * HD + d] = (_Float16)a;
    }
}

// ---------------------------------------------------------------- elementwise GELU f32 -> f16
__global__ __launch_bounds__(256)
void gelu_f32_f16(const float* __restrict__ X, _Float16* __restrict__ Y, size_t n) {
    size_t i = (size_t)blockIdx.x * 256 + threadIdx.x;
    if (i < n) Y[i] = (_Float16)gelu_exact(X[i]);
}

// ---------------------------------------------------------------- row L2-normalize (E=256) f32 -> f16
__global__ __launch_bounds__(256)
void l2norm_rows(const float* __restrict__ X, _Float16* __restrict__ Y)
{
    __shared__ float red[8];
    const size_t row = blockIdx.x;
    const int c = threadIdx.x;
    const float v = X[row * EDIM + c];
    const float ss = block_sum256(v * v, red);
    const float scale = 1.0f / fmaxf(sqrtf(ss), 1e-12f);
    Y[row * EDIM + c] = (_Float16)(v * scale);
}

// ---------------------------------------------------------------- head weights: L2-normalize rows, pad to 1024 rows
__global__ __launch_bounds__(256)
void head_norm_pad(const float* __restrict__ Wt, _Float16* __restrict__ Y)
{
    __shared__ float red[8];
    const int row = blockIdx.x;   // 0..1023
    const int c = threadIdx.x;
    if (row < NCLS) {
        const float v = Wt[(size_t)row * EDIM + c];
        const float ss = block_sum256(v * v, red);
        const float scale = 1.0f / fmaxf(sqrtf(ss), 1e-12f);
        Y[(size_t)row * EDIM + c] = (_Float16)(v * scale);
    } else {
        Y[(size_t)row * EDIM + c] = (_Float16)0.0f;
    }
}

// ================================================================ host side
static inline void launch_cvt(const float* s, _Float16* d, size_t n, hipStream_t st) {
    cvt_f32_f16<<<(unsigned)((n + 255) / 256), 256, 0, st>>>(s, d, n);
}

extern "C" void kernel_launch(void* const* d_in, const int* in_sizes, int n_in,
                              void* d_out, int out_size, void* d_ws, size_t ws_size,
                              hipStream_t stream) {
    const float* x         = (const float*)d_in[0];
    const float* p1_w      = (const float*)d_in[1];
    const float* p1_b      = (const float*)d_in[2];
    const float* pln1_w    = (const float*)d_in[3];
    const float* pln1_b    = (const float*)d_in[4];
    const float* p2_w      = (const float*)d_in[5];
    const float* pln2_w    = (const float*)d_in[6];
    const float* pln2_b    = (const float*)d_in[7];
    const float* in_proj_w = (const float*)d_in[8];
    const float* in_proj_b = (const float*)d_in[9];
    const float* out_proj_w= (const float*)d_in[10];
    const float* out_proj_b= (const float*)d_in[11];
    const float* ln1_w     = (const float*)d_in[12];
    const float* ln1_b     = (const float*)d_in[13];
    const float* lin1_w    = (const float*)d_in[14];
    const float* lin1_b    = (const float*)d_in[15];
    const float* lin2_w    = (const float*)d_in[16];
    const float* lin2_b    = (const float*)d_in[17];
    const float* ln2_w     = (const float*)d_in[18];
    const float* ln2_b     = (const float*)d_in[19];
    const float* head_w    = (const float*)d_in[20];
    float* out = (float*)d_out;

    char* ws = (char*)d_ws;
    // region A: 167,772,160 B : xh f16 [M,2048]; later qkv f32 [M,768] @0 and delta f32 [M,256] @125829120
    _Float16* xh    = (_Float16*)(ws + 0);
    float*    qkv   = (float*)   (ws + 0);
    float*    delta = (float*)   (ws + 125829120);
    // region W (f16 weights), 8,388,608 B
    _Float16* w16      = (_Float16*)(ws + 167772160);
    _Float16* p1_16    = w16 + 0;
    _Float16* p2_16    = w16 + 2097152;
    _Float16* inpj_16  = w16 + 2359296;
    _Float16* outpj_16 = w16 + 2752512;
    _Float16* lin1_16  = w16 + 2883584;
    _Float16* lin2_16  = w16 + 3407872;
    _Float16* headn_16 = w16 + 3932160;
    // region BIG1: f32 [M,1024]
    float*    big1  = (float*)   (ws + 176160768);
    // region BIG2: f16 [M,1024]
    _Float16* big2  = (_Float16*)(ws + 343932928);
    // region RESID: f32 [M,256]
    float*    resid = (float*)   (ws + 427819008);
    // region HF16: f16 [M,256]
    _Float16* hf16  = (_Float16*)(ws + 469762048);
    // region ATT16: f16 [M,256] (attention output; later l2-normalized features)
    _Float16* att16 = (_Float16*)(ws + 490733568);

    // ---- convert inputs/weights to f16
    launch_cvt(x,         xh,      (size_t)MTOT * DIN, stream);
    launch_cvt(p1_w,      p1_16,   (size_t)HID * DIN,  stream);
    launch_cvt(p2_w,      p2_16,   (size_t)EDIM * HID, stream);
    launch_cvt(in_proj_w, inpj_16, (size_t)2 * 3 * EDIM * EDIM, stream);
    launch_cvt(out_proj_w,outpj_16,(size_t)2 * EDIM * EDIM,     stream);
    launch_cvt(lin1_w,    lin1_16, (size_t)2 * HID * EDIM,      stream);
    launch_cvt(lin2_w,    lin2_16, (size_t)2 * EDIM * HID,      stream);

    const dim3 blkG(128);
    // ---- projector GEMM1: [M,2048] x [1024,2048]^T -> big1 f32
    gemm_f16_wmma<<<dim3(HID / 64, MTOT / 64), blkG, 0, stream>>>(
        xh, DIN, p1_16, DIN, p1_b, big1, HID, MTOT, DIN, HID, 1.0f);
    // ---- LN + GELU -> big2 f16
    ln_gelu_f16<<<MTOT, 256, 0, stream>>>(big1, pln1_w, pln1_b, big2, HID);
    // ---- projector GEMM2 (no bias): [M,1024] x [256,1024]^T -> f32 @ region A
    float* h2 = (float*)(ws + 0);
    gemm_f16_wmma<<<dim3(EDIM / 64, MTOT / 64), blkG, 0, stream>>>(
        big2, HID, p2_16, HID, nullptr, h2, EDIM, MTOT, HID, EDIM, 1.0f);
    // ---- LN + positional encoding -> resid f32, hf16
    ln_pe_f32f16<<<MTOT, 256, 0, stream>>>(h2, pln2_w, pln2_b, resid, hf16);

    // ---- 2 transformer encoder layers (post-norm)
    for (int l = 0; l < 2; ++l) {
        // qkv = h @ in_proj^T + b : [M,768]
        gemm_f16_wmma<<<dim3(3 * EDIM / 64, MTOT / 64), blkG, 0, stream>>>(
            hf16, EDIM, inpj_16 + (size_t)l * 3 * EDIM * EDIM, EDIM,
            in_proj_b + (size_t)l * 3 * EDIM, qkv, 3 * EDIM, MTOT, EDIM, 3 * EDIM, 1.0f);
        // fused causal attention -> att16
        attention_fused<<<BATCH * NHEAD, 128, 0, stream>>>(qkv, att16);
        // out_proj -> delta f32
        gemm_f16_wmma<<<dim3(EDIM / 64, MTOT / 64), blkG, 0, stream>>>(
            att16, EDIM, outpj_16 + (size_t)l * EDIM * EDIM, EDIM,
            out_proj_b + (size_t)l * EDIM, delta, EDIM, MTOT, EDIM, EDIM, 1.0f);
        // h = LN(h + attn)
        residual_ln<<<MTOT, 256, 0, stream>>>(resid, delta,
            ln1_w + (size_t)l * EDIM, ln1_b + (size_t)l * EDIM, hf16);
        // ff1: [M,256] x [1024,256]^T -> big1 f32
        gemm_f16_wmma<<<dim3(HID / 64, MTOT / 64), blkG, 0, stream>>>(
            hf16, EDIM, lin1_16 + (size_t)l * HID * EDIM, EDIM,
            lin1_b + (size_t)l * HID, big1, HID, MTOT, EDIM, HID, 1.0f);
        // GELU -> big2 f16
        gelu_f32_f16<<<(unsigned)(((size_t)MTOT * HID + 255) / 256), 256, 0, stream>>>(
            big1, big2, (size_t)MTOT * HID);
        // ff2: [M,1024] x [256,1024]^T -> delta f32
        gemm_f16_wmma<<<dim3(EDIM / 64, MTOT / 64), blkG, 0, stream>>>(
            big2, HID, lin2_16 + (size_t)l * EDIM * HID, HID,
            lin2_b + (size_t)l * EDIM, delta, EDIM, MTOT, HID, EDIM, 1.0f);
        // h = LN(h + ff)
        residual_ln<<<MTOT, 256, 0, stream>>>(resid, delta,
            ln2_w + (size_t)l * EDIM, ln2_b + (size_t)l * EDIM, hf16);
    }

    // ---- cosine head: normalize features & weights, GEMM with alpha = 1/tau
    l2norm_rows<<<MTOT, 256, 0, stream>>>(resid, att16);
    head_norm_pad<<<NCLSP, 256, 0, stream>>>(head_w, headn_16);
    gemm_f16_wmma<<<dim3(NCLSP / 64, MTOT / 64), blkG, 0, stream>>>(
        att16, EDIM, headn_16, EDIM, nullptr, out, NCLS, MTOT, EDIM, NCLS, 10.0f);
}
